// SharedGATNetwork_54443005444824
// MI455X (gfx1250) — matrix operand order repaired and checked by reading, not compiled, not run on hardware
//
#include <hip/hip_runtime.h>
#include <hip/hip_bf16.h>
#include <math.h>
#include <stdint.h>

typedef __attribute__((ext_vector_type(16))) _Float16     v16h;
typedef __attribute__((ext_vector_type(8)))  float        v8f;
typedef __attribute__((ext_vector_type(4)))  unsigned int v4u;
typedef __attribute__((ext_vector_type(8)))  unsigned int v8u;

#define GAT_B 8
#define GAT_N 1024
#define GAT_D 128
#define GAT_K 4
#define GAT_HD 32

// ---------------------------------------------------------------------------
// TDM helper: issue a 2D tensor_load_to_lds (2-group D#, VADDR2/3 = NULL).
// Descriptor packing per CDNA5 ISA 8.3/8.4 (group0 128b, group1 256b).
// ---------------------------------------------------------------------------
__device__ __forceinline__ void tdm_load_2d(unsigned lds_addr, const void* gptr,
                                            unsigned cfg,       // g1[0]: data_size/pad bits
                                            unsigned dim0, unsigned dim1,
                                            unsigned tile0, unsigned tile1,
                                            unsigned stride0) {
  unsigned long long ga = (unsigned long long)(uintptr_t)gptr;
  v4u g0;
  g0.x = 1u;                                   // count=1, user descriptor
  g0.y = lds_addr;                             // lds_addr (bytes)
  g0.z = (unsigned)ga;                         // global_addr[31:0]
  g0.w = (unsigned)((ga >> 32) & 0x01FFFFFFull) | (2u << 30);  // addr[56:32] | type=2
  v8u g1;
  g1[0] = cfg;                                 // wg_mask=0 | data_size | pad ctl
  g1[1] = (dim0 & 0xFFFFu) << 16;              // tensor_dim0[15:0]
  g1[2] = ((dim0 >> 16) & 0xFFFFu) | ((dim1 & 0xFFFFu) << 16);
  g1[3] = ((dim1 >> 16) & 0xFFFFu) | (tile0 << 16);   // tile_dim0
  g1[4] = tile1;                               // tile_dim1 | tile_dim2=0
  g1[5] = stride0;                             // tensor_dim0_stride[31:0]
  g1[6] = 0u;
  g1[7] = 0u;
  asm volatile("tensor_load_to_lds %0, %1" :: "s"(g0), "s"(g1) : "memory");
}

// ---------------------------------------------------------------------------
// Kernel 1: Wh = H @ W   (f32 in, f16 WMMA, f32 out)
// ---------------------------------------------------------------------------
#define HSTR 132
#define WSTR 132

__global__ __launch_bounds__(128) void gat_gemm(const float* __restrict__ H,
                                                const float* __restrict__ W,
                                                float* __restrict__ Wh) {
  __shared__ _Float16 hs[64 * HSTR];
  __shared__ _Float16 wsT[128 * WSTR];   // W transposed: [n][k]
  const int tid = threadIdx.x;
  const int m0  = blockIdx.x * 64;

  for (int idx = tid; idx < GAT_D * GAT_D; idx += 128) {
    int kk = idx & 127, nn = idx >> 7;
    wsT[nn * WSTR + kk] = (_Float16)W[kk * GAT_D + nn];
  }
  for (int idx = tid; idx < 64 * GAT_D; idx += 128) {
    int r = idx >> 7, c = idx & 127;
    hs[r * HSTR + c] = (_Float16)H[(size_t)(m0 + r) * GAT_D + c];
  }
  __syncthreads();

  const int wave = tid >> 5, lane = tid & 31;
  const int lr = lane & 15, hi = lane >> 4;
  const int mrow = wave * 16;

  for (int nt = 0; nt < 8; ++nt) {
    v8f c = {};
    for (int kb = 0; kb < GAT_D; kb += 32) {
      v16h a, b;
      const _Float16* arow = &hs[(mrow + lr) * HSTR + kb + (hi ? 16 : 0)];
      const _Float16* brow = &wsT[(nt * 16 + lr) * WSTR + kb + (hi ? 16 : 0)];
#pragma unroll
      for (int j = 0; j < 8; ++j) { a[j] = arow[j]; a[8 + j] = arow[8 + j]; }
#pragma unroll
      for (int j = 0; j < 16; ++j) { b[j] = brow[j]; }
      c = __builtin_amdgcn_wmma_f32_16x16x32_f16(false, a, false, b,
                                                 (short)0, c, false, false);
    }
#pragma unroll
    for (int v = 0; v < 8; ++v) {
      int row = m0 + mrow + v + (hi ? 8 : 0);
      Wh[(size_t)row * GAT_D + nt * 16 + lr] = c[v];
    }
  }
}

// ---------------------------------------------------------------------------
// Kernel 2: per-(node,head) s,t wave reductions + f16 copy of Wh.
// ---------------------------------------------------------------------------
__global__ __launch_bounds__(256) void gat_st(const float* __restrict__ Wh,
                                              const float* __restrict__ a_src,
                                              const float* __restrict__ a_dst,
                                              float* __restrict__ s,
                                              float* __restrict__ t,
                                              _Float16* __restrict__ Whh) {
  const int gw   = blockIdx.x * 8 + (threadIdx.x >> 5);
  const int lane = threadIdx.x & 31;
  const int node = gw >> 2;
  const int head = gw & 3;
  const size_t idx = (size_t)node * GAT_D + head * GAT_HD + lane;
  const float v = Wh[idx];
  Whh[idx] = (_Float16)v;
  float sv = v * a_src[head * GAT_HD + lane];
  float tv = v * a_dst[head * GAT_HD + lane];
#pragma unroll
  for (int o = 16; o > 0; o >>= 1) {
    sv += __shfl_xor(sv, o, 32);
    tv += __shfl_xor(tv, o, 32);
  }
  if (lane == 0) {
    s[node * GAT_K + head] = sv;
    t[node * GAT_K + head] = tv;
  }
}

// ---------------------------------------------------------------------------
// Kernel 3: masked softmax + WMMA aggregation, TDM-staged, double-buffered.
// Block: 128 threads (4 waves) per (b, k, 64-row i-block). ELU fused.
// ---------------------------------------------------------------------------
#define ATSTR 132   // 128 cols + 4 pad DWORDs (pad produced by TDM pad_enable)

__global__ __launch_bounds__(128) void gat_agg(const float* __restrict__ A,
                                               const float* __restrict__ s,
                                               const float* __restrict__ t,
                                               const _Float16* __restrict__ Whh,
                                               float* __restrict__ out) {
  __shared__ float    t_s[GAT_N];
  __shared__ float    s_s[64];
  __shared__ float    m_s[64];
  __shared__ float    l_s[64];
  __shared__ float    At[2][64 * ATSTR];     // double-buffered A tiles (TDM)
  __shared__ _Float16 Wt[2][128 * 32];       // double-buffered Wh tiles (TDM)

  const int tid = threadIdx.x;
  const int bk  = blockIdx.x;
  const int ib  = bk & 15;
  const int k   = (bk >> 4) & 3;
  const int b   = bk >> 6;
  const int i0  = ib * 64;
  const float* Ab = A + (size_t)b * GAT_N * GAT_N;

  const int wave = tid >> 5, lane = tid & 31;
  const int lr = lane & 15, hi = lane >> 4;
  const int rloc = wave * 16 + lr;
  const int irow = i0 + rloc;

  // TDM descriptor constants
  // A tile: 128x64 f32, row stride 1024 f32; pad 4 DWORDs per 128 DWORDs -> ATSTR
  const unsigned cfgA = (2u << 16) | (1u << 20) | (6u << 22) | (3u << 25);
  // Wh tile: 32x128 f16, row stride 128 halves; contiguous in LDS
  const unsigned cfgW = (1u << 16);
  const unsigned ldsAt[2] = {(unsigned)(uintptr_t)&At[0][0],
                             (unsigned)(uintptr_t)&At[1][0]};
  const unsigned ldsWt[2] = {(unsigned)(uintptr_t)&Wt[0][0],
                             (unsigned)(uintptr_t)&Wt[1][0]};

  // Kick off tile 0 DMAs; they overlap the whole pass-A scan below.
  if (wave == 0) {
    tdm_load_2d(ldsAt[0], Ab + (size_t)i0 * GAT_N, cfgA,
                GAT_N, GAT_N, 128u, 64u, GAT_N);
  } else if (wave == 1) {
    tdm_load_2d(ldsWt[0], Whh + (size_t)b * GAT_N * GAT_D + k * GAT_HD, cfgW,
                4096u, 8192u, 32u, 128u, GAT_D);
  }

  for (int j = tid; j < GAT_N; j += 128) t_s[j] = t[(b * GAT_N + j) * GAT_K + k];
  if (tid < 64) s_s[tid] = s[(b * GAT_N + i0 + tid) * GAT_K + k];
  __syncthreads();

  // ---- Pass A: online softmax stats (2 lanes/row, split by j parity)
  {
    const float si = s_s[rloc];
    float m = -3.0e38f, l = 0.0f;
    for (int jb0 = 0; jb0 < GAT_N; jb0 += 128) {
      __builtin_prefetch(&Ab[(size_t)irow * GAT_N + jb0 + 128 + hi], 0, 0);
      for (int j = jb0 + hi; j < jb0 + 128; j += 2) {
        float a = Ab[(size_t)irow * GAT_N + j];
        if (irow == j && a <= 0.0f) a = 1.0f;        // self-loop fixup
        if (a > 0.0f) {
          float e = si + t_s[j];
          e = e > 0.0f ? e : 0.2f * e;               // leaky relu
          float mn = fmaxf(m, e);
          l = l * __expf(m - mn) + __expf(e - mn);
          m = mn;
        }
      }
    }
    float m2 = __shfl_xor(m, 16, 32);
    float l2 = __shfl_xor(l, 16, 32);
    float mn = fmaxf(m, m2);
    l = l * __expf(m - mn) + l2 * __expf(m2 - mn);
    m = mn;
    if (hi == 0) { m_s[rloc] = m; l_s[rloc] = l; }
  }
  __syncthreads();

  // ---- Pass B: 8 tiles of 128 j-columns, TDM double-buffered
  v8f c0 = {}, c1 = {};
  const float m    = m_s[rloc];
  const float linv = 1.0f / l_s[rloc];
  const float si   = s_s[rloc];

  for (int ti = 0; ti < 8; ++ti) {
    const int jb = ti * 128;
    if (ti + 1 < 8) {
      const int nb = (ti + 1) & 1;
      if (wave == 0) {
        tdm_load_2d(ldsAt[nb], Ab + (size_t)i0 * GAT_N + jb + 128, cfgA,
                    GAT_N, GAT_N, 128u, 64u, GAT_N);
      } else if (wave == 1) {
        tdm_load_2d(ldsWt[nb],
                    Whh + (size_t)(b * GAT_N + jb + 128) * GAT_D + k * GAT_HD,
                    cfgW, 4096u, 8192u, 32u, 128u, GAT_D);
      }
      __builtin_amdgcn_s_wait_tensorcnt(1);   // tile ti complete (in-order)
    } else {
      __builtin_amdgcn_s_wait_tensorcnt(0);
    }
    __syncthreads();

    const float*    Atb = &At[ti & 1][0];
    const _Float16* Wtb = &Wt[ti & 1][0];

    for (int jb2 = 0; jb2 < 128; jb2 += 32) {
      v16h a, b0, b1;
#pragma unroll
      for (int j = 0; j < 16; ++j) {
        int koff = hi * 16 + j;                    // contiguous K index (matches B)
        int jloc = jb2 + koff;
        float av = Atb[rloc * ATSTR + jloc];
        int jg = jb + jloc;
        if (irow == jg && av <= 0.0f) av = 1.0f;
        float alpha = 0.0f;
        if (av > 0.0f) {
          float e = si + t_s[jg];
          e = e > 0.0f ? e : 0.2f * e;
          alpha = __expf(e - m) * linv;
        }
        a[j] = (_Float16)alpha;
      }
#pragma unroll
      for (int j = 0; j < 16; ++j) {
        int kk = jb2 + hi * 16 + j;                // contiguous K index (matches A)
        b0[j] = Wtb[kk * 32 + lr];
        b1[j] = Wtb[kk * 32 + 16 + lr];
      }
      c0 = __builtin_amdgcn_wmma_f32_16x16x32_f16(false, a, false, b0,
                                                  (short)0, c0, false, false);
      c1 = __builtin_amdgcn_wmma_f32_16x16x32_f16(false, a, false, b1,
                                                  (short)0, c1, false, false);
    }
    __syncthreads();   // protect buffer reuse (tile ti+2 overwrites ti&1)
  }

  // ---- Epilogue: ELU + store (C layout: M = v + 8*hi, N = lr)
#pragma unroll
  for (int v = 0; v < 8; ++v) {
    int row = i0 + wave * 16 + v + hi * 8;
    float x0 = c0[v], x1 = c1[v];
    x0 = x0 > 0.0f ? x0 : __expf(x0) - 1.0f;
    x1 = x1 > 0.0f ? x1 : __expf(x1) - 1.0f;
    size_t base = (size_t)(b * GAT_N + row) * GAT_D + k * GAT_HD;
    out[base + lr]      = x0;
    out[base + 16 + lr] = x1;
  }
}

// ---------------------------------------------------------------------------
// Launch
// ---------------------------------------------------------------------------
extern "C" void kernel_launch(void* const* d_in, const int* in_sizes, int n_in,
                              void* d_out, int out_size, void* d_ws, size_t ws_size,
                              hipStream_t stream) {
  const float* X      = (const float*)d_in[0];
  const float* A      = (const float*)d_in[1];
  const float* W0     = (const float*)d_in[2];
  const float* a_src0 = (const float*)d_in[3];
  const float* a_dst0 = (const float*)d_in[4];
  const float* W1     = (const float*)d_in[5];
  const float* a_src1 = (const float*)d_in[6];
  const float* a_dst1 = (const float*)d_in[7];
  float* OUT = (float*)d_out;

  char* ws = (char*)d_ws;
  float*     Wh   = (float*)(ws);                       // 4 MB
  _Float16*  Whh  = (_Float16*)(ws + (4u << 20));       // 2 MB
  float*     sbuf = (float*)(ws + (6u << 20));          // 128 KB
  float*     tbuf = (float*)(ws + (6u << 20) + (128u << 10));
  float*     H1   = (float*)(ws + (6u << 20) + (256u << 10));   // 4 MB

  const int gemm_blocks = (GAT_B * GAT_N) / 64;         // 128
  const int st_blocks   = (GAT_B * GAT_N * GAT_K) / 8;  // 4096
  const int agg_blocks  = GAT_B * GAT_K * (GAT_N / 64); // 512

  // ---- Layer 0
  gat_gemm<<<gemm_blocks, 128, 0, stream>>>(X, W0, Wh);
  gat_st  <<<st_blocks, 256, 0, stream>>>(Wh, a_src0, a_dst0, sbuf, tbuf, Whh);
  gat_agg <<<agg_blocks, 128, 0, stream>>>(A, sbuf, tbuf, Whh, H1);

  // ---- Layer 1
  gat_gemm<<<gemm_blocks, 128, 0, stream>>>(H1, W1, Wh);
  gat_st  <<<st_blocks, 256, 0, stream>>>(Wh, a_src1, a_dst1, sbuf, tbuf, Whh);
  gat_agg <<<agg_blocks, 128, 0, stream>>>(A, sbuf, tbuf, Whh, OUT);
}
